// CoedgeConvLayer_56049323213416
// MI455X (gfx1250) — matrix operand order repaired and checked by reading, not compiled, not run on hardware
//
#include <hip/hip_runtime.h>
#include <stdint.h>

#define D 256
#define BLOCK_M 128
#define THREADS 256

typedef __attribute__((ext_vector_type(16))) __bf16 v16bf;
typedef __attribute__((ext_vector_type(8)))  float  v8f;

struct Q2 { uint4 a, b; };

static __device__ __forceinline__ v16bf make_frag(uint4 x, uint4 y) {
  Q2 q; q.a = x; q.b = y;
  return __builtin_bit_cast(v16bf, q);
}

static __device__ __forceinline__ uint16_t f32_to_bf16_rne(float f) {
  uint32_t u = __builtin_bit_cast(uint32_t, f);
  uint32_t r = u + 0x7FFFu + ((u >> 16) & 1u);
  return (uint16_t)(r >> 16);
}

// ---- Prep: features f32 -> bf16 (8 elems / thread) -------------------------
__global__ void convert_features(const float* __restrict__ in,
                                 uint16_t* __restrict__ out, long long nElems) {
  long long i = ((long long)blockIdx.x * blockDim.x + threadIdx.x) * 8;
  if (i + 8 > nElems) return;
  float4 f0 = *(const float4*)(in + i);
  float4 f1 = *(const float4*)(in + i + 4);
  uint4 o;
  o.x = (uint32_t)f32_to_bf16_rne(f0.x) | ((uint32_t)f32_to_bf16_rne(f0.y) << 16);
  o.y = (uint32_t)f32_to_bf16_rne(f0.z) | ((uint32_t)f32_to_bf16_rne(f0.w) << 16);
  o.z = (uint32_t)f32_to_bf16_rne(f1.x) | ((uint32_t)f32_to_bf16_rne(f1.y) << 16);
  o.w = (uint32_t)f32_to_bf16_rne(f1.z) | ((uint32_t)f32_to_bf16_rne(f1.w) << 16);
  *(uint4*)(out + i) = o;
}

// ---- Prep: W[k][n] f32 -> WT[s][n][k] bf16 (transposed, concatenated) ------
__global__ void prep_weights(const float* __restrict__ Ws, const float* __restrict__ Wn,
                             const float* __restrict__ Wp, const float* __restrict__ Wm,
                             uint16_t* __restrict__ WT) {
  int g = blockIdx.x * blockDim.x + threadIdx.x;   // 0 .. 4*256*256-1
  int s    = g >> 16;
  int k    = (g >> 8) & 255;
  int ncol = g & 255;                              // coalesced read over ncol
  const float* W = (s == 0) ? Ws : (s == 1) ? Wn : (s == 2) ? Wp : Wm;
  WT[((size_t)s * D + ncol) * D + k] = f32_to_bf16_rne(W[k * D + ncol]);
}

// ---- Prep: summed bias -----------------------------------------------------
__global__ void prep_bias(const float* __restrict__ b0, const float* __restrict__ b1,
                          const float* __restrict__ b2, const float* __restrict__ b3,
                          float* __restrict__ bsum) {
  int i = threadIdx.x;
  bsum[i] = b0[i] + b1[i] + b2[i] + b3[i];
}

// ---- Main fused gather + GEMM + bias + relu --------------------------------
// Each wave: 16 output rows x 256 cols (16 WMMA C-tiles). Block: 8 waves.
// K = 4 sources x 256 = 1024, chunks of 32 (bf16 WMMA).
// Pipelining: LDS slab double-buffered across chunks; B fragments
// double-buffered in registers within a chunk.
__global__ void __launch_bounds__(THREADS) coedge_gemm(
    const uint16_t* __restrict__ hb,   // [n][256] bf16
    const uint16_t* __restrict__ WT,   // [4][256(n)][256(k)] bf16
    const float* __restrict__ bsum,    // [256]
    const int* __restrict__ nidx, const int* __restrict__ pidx,
    const int* __restrict__ midx,
    float* __restrict__ out, int n) {
  // Two B slabs (one 32-K chunk each): 2 x 256 cols x 32 k x bf16 = 32 KB
  __shared__ __align__(16) uint16_t sB[2][D * 32];

  const int tid      = threadIdx.x;
  const int wave     = tid >> 5;
  const int lane     = tid & 31;
  const int lane16   = lane & 15;
  const int laneHalf = lane >> 4;

  const int rowBase = blockIdx.x * BLOCK_M + wave * 16;
  int mA = rowBase + lane16;
  if (mA > n - 1) mA = n - 1;   // clamp ragged tail (stores are guarded)

  const uint16_t* a0 = hb + (size_t)mA * D;
  const uint16_t* a1 = hb + (size_t)nidx[mA] * D;
  const uint16_t* a2 = hb + (size_t)pidx[mA] * D;
  const uint16_t* a3 = hb + (size_t)midx[mA] * D;

  v8f acc[16] = {};

  // Per-lane B fragment base offset (elements) within a slab:
  // col = nt*16 + lane16, 16 contiguous K at laneHalf*16; nt stride = 512.
  const int bOff = lane16 * 32 + laneHalf * 16;

  // ---- prologue: fetch chunk 0 (s=0, k0=0) and stage into sB[0] ----
  uint4 aCur0, aCur1;
  {
    const uint4* src = (const uint4*)(WT + (size_t)tid * D);
    uint4 b0 = src[0], b1 = src[1], b2 = src[2], b3 = src[3];
    const uint16_t* ap = a0 + laneHalf * 16;
    aCur0 = *(const uint4*)ap;
    aCur1 = *(const uint4*)(ap + 8);
    uint4* dst = (uint4*)(&sB[0][tid * 32]);
    dst[0] = b0; dst[1] = b1; dst[2] = b2; dst[3] = b3;
  }
  __syncthreads();

#pragma unroll 1
  for (int c = 0; c < 32; ++c) {
    const int cur = c & 1;

    // ---- issue prefetch for chunk c+1 (loads overlap the WMMA chain) ----
    uint4 nb0 = {}, nb1 = {}, nb2 = {}, nb3 = {}, na0 = {}, na1 = {};
    if (c < 31) {
      const int cn  = c + 1;
      const int sn  = cn >> 3;          // source of next chunk
      const int k0n = (cn & 7) * 32;    // k offset within source
      const uint4* src =
          (const uint4*)(WT + (size_t)sn * (D * D) + (size_t)tid * D + k0n);
      nb0 = src[0]; nb1 = src[1]; nb2 = src[2]; nb3 = src[3];
      const uint16_t* abn = (sn == 0) ? a0 : (sn == 1) ? a1 : (sn == 2) ? a2 : a3;
      const uint16_t* ap  = abn + k0n + laneHalf * 16;
      na0 = *(const uint4*)ap;
      na1 = *(const uint4*)(ap + 8);
    }

    // ---- compute chunk c from sB[cur]; B frags double-buffered in regs ----
    v16bf aF = make_frag(aCur0, aCur1);
    const uint16_t* sbase = &sB[cur][0] + bOff;

    const uint4* bp0 = (const uint4*)sbase;
    v16bf bF = make_frag(bp0[0], bp0[1]);
#pragma unroll
    for (int nt = 0; nt < 16; ++nt) {
      if (nt < 15) {
        // issue next fragment's ds_loads before consuming the current one
        const uint4* bp = (const uint4*)(sbase + (nt + 1) * 512);
        v16bf bN = make_frag(bp[0], bp[1]);
        acc[nt] = __builtin_amdgcn_wmma_f32_16x16x32_bf16(
            false, aF, false, bF, (short)0, acc[nt], false, false);
        bF = bN;
      } else {
        acc[nt] = __builtin_amdgcn_wmma_f32_16x16x32_bf16(
            false, aF, false, bF, (short)0, acc[nt], false, false);
      }
    }

    // ---- stage prefetched slab into the other buffer ----
    if (c < 31) {
      uint4* dst = (uint4*)(&sB[1 - cur][tid * 32]);
      dst[0] = nb0; dst[1] = nb1; dst[2] = nb2; dst[3] = nb3;
      __syncthreads();
      aCur0 = na0; aCur1 = na1;
    }
  }

  // Epilogue: C tile element (m = r + 8*laneHalf, n = lane16) per VGPR r
  const int mTop = rowBase + laneHalf * 8;
#pragma unroll
  for (int nt = 0; nt < 16; ++nt) {
    const int ncol = nt * 16 + lane16;
    const float bs = bsum[ncol];
#pragma unroll
    for (int r = 0; r < 8; ++r) {
      const int m = mTop + r;
      if (m < n) {
        float v = acc[nt][r] + bs;
        out[(size_t)m * D + ncol] = v > 0.0f ? v : 0.0f;
      }
    }
  }
}

extern "C" void kernel_launch(void* const* d_in, const int* in_sizes, int n_in,
                              void* d_out, int out_size, void* d_ws, size_t ws_size,
                              hipStream_t stream) {
  (void)n_in; (void)out_size; (void)ws_size;
  const float* features = (const float*)d_in[0];
  const int*   nidx     = (const int*)d_in[1];
  const int*   pidx     = (const int*)d_in[2];
  const int*   midx     = (const int*)d_in[3];
  const float* W_self   = (const float*)d_in[4];
  const float* b_self   = (const float*)d_in[5];
  const float* W_next   = (const float*)d_in[6];
  const float* b_next   = (const float*)d_in[7];
  const float* W_prev   = (const float*)d_in[8];
  const float* b_prev   = (const float*)d_in[9];
  const float* W_mate   = (const float*)d_in[10];
  const float* b_mate   = (const float*)d_in[11];

  const int n = in_sizes[1];                 // number of coedges
  const long long elems = (long long)n * D;

  // Workspace layout: bf16 features | bf16 WT (4*256*256) | f32 bias sum
  uint16_t* hb   = (uint16_t*)d_ws;
  uint16_t* WT   = hb + (size_t)n * D;
  float*    bsum = (float*)(WT + 4 * D * D);

  const int cvtBlocks = (int)((elems / 8 + 255) / 256);
  convert_features<<<cvtBlocks, 256, 0, stream>>>(features, hb, elems);
  prep_weights<<<(4 * D * D) / 256, 256, 0, stream>>>(W_self, W_next, W_prev, W_mate, WT);
  prep_bias<<<1, 256, 0, stream>>>(b_self, b_next, b_prev, b_mate, bsum);

  const int gblocks = (n + BLOCK_M - 1) / BLOCK_M;
  coedge_gemm<<<gblocks, THREADS, 0, stream>>>(hb, WT, bsum, nidx, pidx, midx,
                                               (float*)d_out, n);
}
